// CausalSelfAttention_44770739093968
// MI455X (gfx1250) — matrix operand order repaired and checked by reading, not compiled, not run on hardware
//
#include <hip/hip_runtime.h>

typedef __attribute__((ext_vector_type(16))) _Float16 v16h;
typedef __attribute__((ext_vector_type(8)))  _Float16 v8h;
typedef __attribute__((ext_vector_type(4)))  _Float16 v4h;
typedef __attribute__((ext_vector_type(8)))  float    v8f;
typedef __attribute__((ext_vector_type(4)))  unsigned v4u;
typedef __attribute__((ext_vector_type(8)))  int      v8i;
typedef __attribute__((ext_vector_type(4)))  int      v4i;

#define WMMA16(a,b,c) __builtin_amdgcn_wmma_f32_16x16x32_f16(false,(a),false,(b),(short)0,(c),false,false)

namespace {
constexpr int B_ = 4, T_ = 2048, C_ = 1024, H_ = 16, D_ = 64;
constexpr int M_ = B_ * T_;        // 8192 rows
constexpr int NQKV_ = 3 * C_;      // 3072
constexpr int LDA = 40;            // 32 f16 + 4-dword pad (matches TDM pad descriptor)
constexpr unsigned SB_OFF = 128 * LDA * 2;            // 10240: B tile after A tile
constexpr unsigned BUF_STRIDE = SB_OFF + 64 * LDA * 2; // 15360: one ping-pong buffer
constexpr unsigned SMEM_BYTES = 2 * BUF_STRIDE;        // 30720
}

// ---------- Tensor Data Mover descriptors (CDNA5 ISA) ----------
// Group 0: [1:0]=count=1, [63:32]=lds_addr, [120:64]=global_addr, [127:126]=type=2
static __device__ inline v4u tdm_g0(const void* gaddr, unsigned lds_off) {
  unsigned long long a = (unsigned long long)gaddr;
  v4u g;
  g[0] = 1u;
  g[1] = lds_off;
  g[2] = (unsigned)a;
  g[3] = (unsigned)(a >> 32) | (2u << 30);
  return g;
}
// Group 1: data_size=2B(code1)@16, pad_enable@20, pad_interval=16dw(code3)@22,
// pad_amount=4dw(code3)@25; tensor_dim0@[79:48], tensor_dim1@[111:80],
// tile_dim0@[127:112], tile_dim1@[143:128], tensor_dim0_stride@[207:160].
static __device__ inline v8i tdm_g1(unsigned dim0, unsigned dim1, unsigned stride0,
                                    unsigned tile0, unsigned tile1) {
  v8i g;
  g[0] = (int)((1u << 16) | (1u << 20) | (3u << 22) | (3u << 25));
  g[1] = (int)((dim0 & 0xFFFFu) << 16);
  g[2] = (int)((dim0 >> 16) | ((dim1 & 0xFFFFu) << 16));
  g[3] = (int)((dim1 >> 16) | (tile0 << 16));
  g[4] = (int)tile1;      // tile_dim2 = 0 (2-D)
  g[5] = (int)stride0;    // stride low 32 (elements)
  g[6] = 0;
  g[7] = 0;
  return g;
}
static __device__ inline void tdm_load(v4u g0, v8i g1) {
  v4i z4 = {0, 0, 0, 0};
#if defined(__clang_major__) && (__clang_major__ >= 23)
  v8i z8 = {0, 0, 0, 0, 0, 0, 0, 0};
  __builtin_amdgcn_tensor_load_to_lds(g0, g1, z4, z4, z8, 0);
#else
  __builtin_amdgcn_tensor_load_to_lds(g0, g1, z4, z4, 0);
#endif
}

// ---------------- fp32 -> fp16 convert (vectorized x4) ----------------
__global__ __launch_bounds__(256) void cvt_f32_f16(const float* __restrict__ in,
                                                   _Float16* __restrict__ out, int n4) {
  int i = blockIdx.x * 256 + threadIdx.x;
  if (i < n4) {
    float4 f = ((const float4*)in)[i];
    v4h o;
    o[0] = (_Float16)f.x; o[1] = (_Float16)f.y;
    o[2] = (_Float16)f.z; o[3] = (_Float16)f.w;
    ((v4h*)out)[i] = o;
  }
}

// ------------- fp32 [R,Ccols] -> fp16 transposed [Ccols,R] -------------
__global__ __launch_bounds__(256) void transpose_f32_f16(const float* __restrict__ in,
                                                         _Float16* __restrict__ out,
                                                         int R, int Ccols) {
  __shared__ float tile[32][33];
  int bx = blockIdx.x * 32;
  int by = blockIdx.y * 32;
  int tx = threadIdx.x & 31, ty = threadIdx.x >> 5;
#pragma unroll
  for (int r = ty; r < 32; r += 8)
    tile[r][tx] = in[(size_t)(by + r) * Ccols + bx + tx];
  __syncthreads();
#pragma unroll
  for (int r = ty; r < 32; r += 8)
    out[(size_t)(bx + r) * R + by + tx] = (_Float16)tile[tx][r];
}

// ---- shared GEMM core: TDM double-buffered 128x64 tile, 8 waves, K-step 32 ----
// Computes acc[4] (16x16 tiles) for this wave; caller does the epilogue.
static __device__ inline void gemm_core(const _Float16* A, const _Float16* Bt,
                                        unsigned bRows, int m0, int n0,
                                        int wave, int l16, int hf,
                                        char* smem, v8f acc[4]) {
  if (wave == 0) {  // prologue: DMA tile 0 into buffer 0
    tdm_load(tdm_g0(A + (size_t)m0 * C_, 0u), tdm_g1(C_, M_, C_, 32, 128));
    tdm_load(tdm_g0(Bt + (size_t)n0 * C_, SB_OFF), tdm_g1(C_, bRows, C_, 32, 64));
    __builtin_amdgcn_s_wait_tensorcnt(0);
  }
  __syncthreads();

  for (int k0 = 0; k0 < C_; k0 += 32) {
    const int p = (k0 >> 5) & 1;
    if (k0 + 32 < C_ && wave == 0) {  // async: DMA next tile into other buffer
      unsigned dst = (1 - p) * BUF_STRIDE;
      tdm_load(tdm_g0(A + (size_t)m0 * C_ + k0 + 32, dst),
               tdm_g1(C_, M_, C_, 32, 128));
      tdm_load(tdm_g0(Bt + (size_t)n0 * C_ + k0 + 32, dst + SB_OFF),
               tdm_g1(C_, bRows, C_, 32, 64));
    }

    const _Float16* sA = (const _Float16*)(smem + p * BUF_STRIDE);
    const _Float16* sB = sA + SB_OFF / 2;

    v16h af;
    {
      int row = wave * 16 + l16;
      v8h lo = *(const v8h*)&sA[row * LDA + 8 * hf];
      v8h hi = *(const v8h*)&sA[row * LDA + 8 * hf + 16];
#pragma unroll
      for (int e = 0; e < 8; ++e) { af[e] = lo[e]; af[e + 8] = hi[e]; }
    }
    v16h bf[4];
#pragma unroll
    for (int ct = 0; ct < 4; ++ct) {  // hoist all B-frag loads: one clause, one wait
      int n = ct * 16 + l16;
      v8h lo = *(const v8h*)&sB[n * LDA + 8 * hf];
      v8h hi = *(const v8h*)&sB[n * LDA + 8 * hf + 16];
#pragma unroll
      for (int e = 0; e < 8; ++e) { bf[ct][e] = lo[e]; bf[ct][e + 8] = hi[e]; }
    }
#pragma unroll
    for (int ct = 0; ct < 4; ++ct)    // back-to-back WMMAs, no intervening waits
      acc[ct] = WMMA16(af, bf[ct], acc[ct]);

    if (wave == 0) __builtin_amdgcn_s_wait_tensorcnt(0);  // drain DMA
    __syncthreads();                  // everyone done reading buf p; next tile ready
  }
}

// ---------------- QKV GEMM: [8192,1024] x Wt[3072,1024] ----------------
// Q (scaled 1/8), K -> f16 [B,H,T,D]; V -> transposed f16 [B,H,D,T].
__global__ __launch_bounds__(256) void gemm_qkv(const _Float16* __restrict__ X,
                                                const _Float16* __restrict__ Wt,
                                                _Float16* __restrict__ Qo,
                                                _Float16* __restrict__ Ko,
                                                _Float16* __restrict__ Vt) {
  extern __shared__ __align__(16) char smem[];
  const int tid = threadIdx.x;
  const int lane = tid & 31, wave = tid >> 5;
  const int l16 = lane & 15, hf = (lane >> 4) & 1;
  const int m0 = blockIdx.x * 128, n0 = blockIdx.y * 64;

  v8f acc[4] = {};
  gemm_core(X, Wt, NQKV_, m0, n0, wave, l16, hf, smem, acc);

  const int region = n0 >> 10;                 // 0=q 1=k 2=v
  const int mbase  = m0 + wave * 16 + 8 * hf;
  const int b = mbase >> 11, t0 = mbase & (T_ - 1);
#pragma unroll
  for (int ct = 0; ct < 4; ++ct) {
    int within = (n0 + ct * 16 + l16) & (C_ - 1);
    int hh = within >> 6, d = within & 63;
    if (region == 2) {                         // V transposed: contiguous t, packed store
      v8h pack;
#pragma unroll
      for (int r = 0; r < 8; ++r) pack[r] = (_Float16)acc[ct][r];
      *(v8h*)&Vt[((size_t)(b * H_ + hh) * D_ + d) * T_ + t0] = pack;
    } else {
      _Float16* dst = (region == 0) ? Qo : Ko;
      float sc = (region == 0) ? 0.125f : 1.0f;  // fold 1/sqrt(64) into Q
#pragma unroll
      for (int r = 0; r < 8; ++r)
        dst[((size_t)(b * H_ + hh) * T_ + (t0 + r)) * D_ + d] = (_Float16)(acc[ct][r] * sc);
    }
  }
}

// ---------------- Flash attention: one independent wave per 16 q-rows ----------------
__global__ __launch_bounds__(128) void flash_attn(const _Float16* __restrict__ Q,
                                                  const _Float16* __restrict__ K,
                                                  const _Float16* __restrict__ Vt,
                                                  _Float16* __restrict__ Y) {
  const int tid = threadIdx.x;
  const int lane = tid & 31, wave = tid >> 5;
  const int l16 = lane & 15, hf = (lane >> 4) & 1;
  const int bh = blockIdx.y;
  const int q0 = blockIdx.x * 64 + wave * 16;
  const _Float16* Qb = Q  + (size_t)bh * T_ * D_;
  const _Float16* Kb = K  + (size_t)bh * T_ * D_;
  const _Float16* Vb = Vt + (size_t)bh * D_ * T_;

  v16h qf[2];                                  // Q^T B-fragments
#pragma unroll
  for (int c = 0; c < 2; ++c)
    qf[c] = *(const v16h*)&Qb[(size_t)(q0 + l16) * D_ + 32 * c + 16 * hf];

  v8f o[4] = {};
  float mrow = -1e30f, lrow = 0.f;
  const int myq = q0 + l16;
  const int kend = ((q0 + 15) >> 5) << 5;

  for (int k0 = 0; k0 <= kend; k0 += 32) {
    v8f s[2] = {};                             // S^T, two 16-key chunks
#pragma unroll
    for (int ch = 0; ch < 2; ++ch) {
#pragma unroll
      for (int c = 0; c < 2; ++c) {
        const _Float16* kp = &Kb[(size_t)(k0 + 16 * ch + l16) * D_ + 32 * c + 16 * hf];
        v8h lo = *(const v8h*)kp;
        v8h hi = *(const v8h*)(kp + 8);
        v16h af;
#pragma unroll
        for (int e = 0; e < 8; ++e) { af[e] = lo[e]; af[e + 8] = hi[e]; }
        s[ch] = WMMA16(af, qf[c], s[ch]);
      }
    }
    float vmax = -1e30f;
#pragma unroll
    for (int ch = 0; ch < 2; ++ch)
#pragma unroll
      for (int r = 0; r < 8; ++r) {
        int key = k0 + 16 * ch + 8 * hf + r;
        float sv = (key <= myq) ? s[ch][r] : -1e30f;
        s[ch][r] = sv;
        vmax = fmaxf(vmax, sv);
      }
    vmax = fmaxf(vmax, __shfl_xor(vmax, 16, 32));
    float mnew  = fmaxf(mrow, vmax);
    float alpha = __expf(mrow - mnew);
    mrow = mnew;

    float psum = 0.f;
    v16h pf;                                   // exp() lands directly in A-frag layout
#pragma unroll
    for (int ch = 0; ch < 2; ++ch)
#pragma unroll
      for (int r = 0; r < 8; ++r) {
        float p = __expf(s[ch][r] - mnew);
        psum += p;
        pf[ch * 8 + r] = (_Float16)p;
      }
    psum += __shfl_xor(psum, 16, 32);
    lrow = lrow * alpha + psum;

    float asc[8];
#pragma unroll
    for (int r = 0; r < 8; ++r) asc[r] = __shfl(alpha, r + 8 * hf, 32);
#pragma unroll
    for (int n = 0; n < 4; ++n)
#pragma unroll
      for (int r = 0; r < 8; ++r) o[n][r] *= asc[r];

#pragma unroll
    for (int n = 0; n < 4; ++n) {              // P(16x32) @ V(32x16), V^T contiguous
      const _Float16* vp = &Vb[(size_t)(16 * n + l16) * T_ + k0 + 8 * hf];
      v8h vlo = *(const v8h*)vp;
      v8h vhi = *(const v8h*)(vp + 16);
      v16h vf;
#pragma unroll
      for (int e = 0; e < 8; ++e) { vf[e] = vlo[e]; vf[e + 8] = vhi[e]; }
      o[n] = WMMA16(pf, vf, o[n]);
    }
  }

  float inv = 1.0f / lrow;
  float isc[8];
#pragma unroll
  for (int r = 0; r < 8; ++r) isc[r] = __shfl(inv, r + 8 * hf, 32);

  const int b = bh >> 4, hh = bh & 15;
#pragma unroll
  for (int n = 0; n < 4; ++n)
#pragma unroll
    for (int r = 0; r < 8; ++r) {
      int t = q0 + 8 * hf + r;
      int col = hh * 64 + 16 * n + l16;
      Y[((size_t)(b * T_ + t)) * C_ + col] = (_Float16)(o[n][r] * isc[r]);
    }
}

// ---------------- Output projection: [8192,1024] x Wt[1024,1024] -> f32 ----------------
__global__ __launch_bounds__(256) void gemm_proj(const _Float16* __restrict__ Yh,
                                                 const _Float16* __restrict__ Wt,
                                                 float* __restrict__ Out) {
  extern __shared__ __align__(16) char smem[];
  const int tid = threadIdx.x;
  const int lane = tid & 31, wave = tid >> 5;
  const int l16 = lane & 15, hf = (lane >> 4) & 1;
  const int m0 = blockIdx.x * 128, n0 = blockIdx.y * 64;

  v8f acc[4] = {};
  gemm_core(Yh, Wt, C_, m0, n0, wave, l16, hf, smem, acc);

  const int mbase = m0 + wave * 16 + 8 * hf;
#pragma unroll
  for (int ct = 0; ct < 4; ++ct) {
    int n = n0 + ct * 16 + l16;
#pragma unroll
    for (int r = 0; r < 8; ++r)
      Out[(size_t)(mbase + r) * C_ + n] = acc[ct][r];
  }
}

extern "C" void kernel_launch(void* const* d_in, const int* in_sizes, int n_in,
                              void* d_out, int out_size, void* d_ws, size_t ws_size,
                              hipStream_t stream) {
  const float* x     = (const float*)d_in[0];
  const float* wqkv  = (const float*)d_in[1];
  const float* wproj = (const float*)d_in[2];
  float* out = (float*)d_out;

  char* ws = (char*)d_ws;
  _Float16* xh  = (_Float16*)ws; ws += (size_t)M_ * C_ * 2;       // 16 MB
  _Float16* wtq = (_Float16*)ws; ws += (size_t)C_ * NQKV_ * 2;    //  6 MB (W_qkv^T)
  _Float16* wtp = (_Float16*)ws; ws += (size_t)C_ * C_ * 2;       //  2 MB (W_proj^T)
  _Float16* qh  = (_Float16*)ws; ws += (size_t)M_ * C_ * 2;       // 16 MB
  _Float16* kh  = (_Float16*)ws; ws += (size_t)M_ * C_ * 2;       // 16 MB
  _Float16* vth = (_Float16*)ws; ws += (size_t)M_ * C_ * 2;       // 16 MB
  _Float16* yh  = (_Float16*)ws; ws += (size_t)M_ * C_ * 2;       // 16 MB

  int n4x = (M_ * C_) / 4;
  cvt_f32_f16<<<(n4x + 255) / 256, 256, 0, stream>>>(x, xh, n4x);
  transpose_f32_f16<<<dim3(NQKV_ / 32, C_ / 32), 256, 0, stream>>>(wqkv, wtq, C_, NQKV_);
  transpose_f32_f16<<<dim3(C_ / 32, C_ / 32), 256, 0, stream>>>(wproj, wtp, C_, C_);

  gemm_qkv<<<dim3(M_ / 128, NQKV_ / 64), 256, SMEM_BYTES, stream>>>(xh, wtq, qh, kh, vth);
  flash_attn<<<dim3(T_ / 64, B_ * H_), 128, 0, stream>>>(qh, kh, vth, yh);
  gemm_proj<<<dim3(M_ / 128, C_ / 64), 256, SMEM_BYTES, stream>>>(yh, wtp, out);
}
